// SparseMoE_Text_9517647528396
// MI455X (gfx1250) — matrix operand order repaired and checked by reading, not compiled, not run on hardware
//
#include <hip/hip_runtime.h>
#include <hip/hip_bf16.h>
#include <math.h>

// ---------------- problem constants (from reference) ----------------
#define N_TOK 16384      // B*T
#define C_DIM 1024
#define H_DIM 4096
#define E_NUM 8
#define MT    32         // tokens per FFN tile-block
#define HC    64         // H chunk (columns of h produced per iteration)

// ---------------- workspace layout (bytes) ----------------
// cnt[E] | tok[E][N] | prb[E][N] | x_bf16[N][C] | w1t_bf16[E][H][C] | w2t_bf16[E][C][H]
static constexpr size_t OFF_CNT = 0;
static constexpr size_t OFF_TOK = 256;
static constexpr size_t OFF_PRB = OFF_TOK + (size_t)E_NUM * N_TOK * 4;
static constexpr size_t OFF_XB  = OFF_PRB + (size_t)E_NUM * N_TOK * 4;
static constexpr size_t OFF_W1T = OFF_XB  + (size_t)N_TOK * C_DIM * 2;
static constexpr size_t OFF_W2T = OFF_W1T + (size_t)E_NUM * C_DIM * H_DIM * 2;
static constexpr size_t WS_NEED = OFF_W2T + (size_t)E_NUM * C_DIM * H_DIM * 2; // ~161 MB

// ---------------- types for WMMA ----------------
typedef __attribute__((ext_vector_type(16))) __bf16        v16bf;
typedef __attribute__((ext_vector_type(8)))  float         v8f;
typedef __attribute__((ext_vector_type(8)))  unsigned int  v8u;

__device__ __forceinline__ unsigned short bf16_rne(float f) {
  unsigned int u = __float_as_uint(f);
  unsigned int r = u + 0x7FFFu + ((u >> 16) & 1u);
  return (unsigned short)(r >> 16);
}

// B operand: 16 contiguous bf16 (K = base..base+15 for this lane's half)
__device__ __forceinline__ v16bf ld_b_contig(const unsigned short* p) {
  const uint4* q = (const uint4*)p;
  uint4 a = q[0], b = q[1];
  v8u v; v[0]=a.x; v[1]=a.y; v[2]=a.z; v[3]=a.w; v[4]=b.x; v[5]=b.y; v[6]=b.z; v[7]=b.w;
  return __builtin_bit_cast(v16bf, v);
}

// A operand: two 16B chunks, halves 0-7 at p[0..7] (K=8h+0..7), halves 8-15 at p[16..23] (K=8h+16..23)
__device__ __forceinline__ v16bf ld_a_split(const unsigned short* p) {
  uint4 a = *(const uint4*)p;
  uint4 b = *(const uint4*)(p + 16);
  v8u v; v[0]=a.x; v[1]=a.y; v[2]=a.z; v[3]=a.w; v[4]=b.x; v[5]=b.y; v[6]=b.z; v[7]=b.w;
  return __builtin_bit_cast(v16bf, v);
}

// ---------------- kernel 1: zero output + expert counters ----------------
__global__ void moe_zero_kernel(float* __restrict__ out, int* __restrict__ cnt, size_t n) {
  size_t i = (size_t)blockIdx.x * blockDim.x + threadIdx.x;
  size_t stride = (size_t)gridDim.x * blockDim.x;
  for (; i < n; i += stride) out[i] = 0.f;
  if (blockIdx.x == 0 && threadIdx.x < E_NUM) cnt[threadIdx.x] = 0;
}

// ---------------- kernel 2: f32 -> bf16 elementwise (x) ----------------
__global__ void moe_cvt_kernel(const float* __restrict__ in, unsigned short* __restrict__ out, size_t n) {
  size_t i = (size_t)blockIdx.x * blockDim.x + threadIdx.x;
  size_t stride = (size_t)gridDim.x * blockDim.x;
  for (; i < n; i += stride) out[i] = bf16_rne(in[i]);
}

// ---------------- kernel 3: transpose + convert (weights) ----------------
// in: [E][R][Cc] f32 -> out: [E][Cc][R] bf16 ; 32x32 LDS tiles
__global__ __launch_bounds__(256) void moe_transpose_cvt_kernel(
    const float* __restrict__ in, unsigned short* __restrict__ out, int R, int Cc) {
  __shared__ unsigned short tile[32][33];
  int e  = blockIdx.z;
  int r0 = blockIdx.y * 32, c0 = blockIdx.x * 32;
  const float* pin = in + (size_t)e * R * Cc;
  unsigned short* pout = out + (size_t)e * R * Cc;
  int lx = threadIdx.x & 31, ly = threadIdx.x >> 5;
  #pragma unroll
  for (int i = 0; i < 32; i += 8)
    tile[ly + i][lx] = bf16_rne(pin[(size_t)(r0 + ly + i) * Cc + c0 + lx]);
  __syncthreads();
  #pragma unroll
  for (int i = 0; i < 32; i += 8)
    pout[(size_t)(c0 + ly + i) * R + r0 + lx] = tile[lx][ly + i];
}

// ---------------- kernel 4: routing (one wave32 per token) ----------------
__global__ __launch_bounds__(256) void moe_route_kernel(
    const float* __restrict__ x,  const float* __restrict__ eps,
    const float* __restrict__ rw, const float* __restrict__ rb,
    const float* __restrict__ nw, const float* __restrict__ nb,
    int* __restrict__ cnt, int* __restrict__ tokl, float* __restrict__ prbl) {
  int lane = threadIdx.x & 31;
  int wave = threadIdx.x >> 5;
  int token = blockIdx.x * 8 + wave;
  float aR[E_NUM] = {0.f,0.f,0.f,0.f,0.f,0.f,0.f,0.f};
  float aN[E_NUM] = {0.f,0.f,0.f,0.f,0.f,0.f,0.f,0.f};
  const float* xr = x + (size_t)token * C_DIM;
  for (int i = 0; i < C_DIM / 32; ++i) {
    int c = i * 32 + lane;
    float xv = xr[c];
    const float* prw = rw + c * E_NUM;
    const float* pnw = nw + c * E_NUM;
    #pragma unroll
    for (int e = 0; e < E_NUM; ++e) {
      aR[e] = fmaf(xv, prw[e], aR[e]);
      aN[e] = fmaf(xv, pnw[e], aN[e]);
    }
  }
  #pragma unroll
  for (int off = 16; off > 0; off >>= 1) {
    #pragma unroll
    for (int e = 0; e < E_NUM; ++e) {
      aR[e] += __shfl_xor(aR[e], off, 32);
      aN[e] += __shfl_xor(aN[e], off, 32);
    }
  }
  if (lane == 0) {
    float noisy[E_NUM];
    #pragma unroll
    for (int e = 0; e < E_NUM; ++e) {
      float lg = aR[e] + rb[e];
      float z  = aN[e] + nb[e];
      float sp = (z > 30.f) ? z : log1pf(__expf(z));   // softplus
      noisy[e] = lg + eps[(size_t)token * E_NUM + e] * sp;
    }
    // top-2, ties -> lower index (matches jax.lax.top_k)
    int i1 = 0; float v1 = noisy[0];
    #pragma unroll
    for (int e = 1; e < E_NUM; ++e) if (noisy[e] > v1) { v1 = noisy[e]; i1 = e; }
    int i2 = -1; float v2 = -3.4e38f;
    #pragma unroll
    for (int e = 0; e < E_NUM; ++e) if (e != i1 && noisy[e] > v2) { v2 = noisy[e]; i2 = e; }
    // 2-way softmax over selected noisy logits
    float p2 = __expf(v2 - v1);
    float s  = 1.f + p2;
    float q1 = 1.f / s, q2 = p2 / s;
    int c1 = atomicAdd(&cnt[i1], 1);
    tokl[i1 * N_TOK + c1] = token; prbl[i1 * N_TOK + c1] = q1;
    int c2 = atomicAdd(&cnt[i2], 1);
    tokl[i2 * N_TOK + c2] = token; prbl[i2 * N_TOK + c2] = q2;
  }
}

// ---------------- kernel 5: fused expert FFN with bf16 WMMA ----------------
// block = 256 threads (8 waves); one block computes 32 gathered tokens for one expert.
__global__ __launch_bounds__(256) void moe_ffn_kernel(
    const float* __restrict__ b1, const float* __restrict__ b2,
    const unsigned short* __restrict__ xb,
    const unsigned short* __restrict__ w1t,   // [E][H][C] bf16
    const unsigned short* __restrict__ w2t,   // [E][C][H] bf16
    const int* __restrict__ cnt, const int* __restrict__ tokl,
    const float* __restrict__ prbl, float* __restrict__ out) {
  int e     = blockIdx.y;
  int count = cnt[e];
  int base  = blockIdx.x * MT;
  if (base >= count) return;                       // uniform per block

  __shared__ unsigned short Xs[MT][C_DIM];         // 64 KB  (gathered tokens, bf16)
  __shared__ unsigned short Hs[MT][HC];            // 4 KB   (relu'd hidden chunk, bf16)
  __shared__ int   tok_s[MT];
  __shared__ float prb_s[MT];

  int tid  = threadIdx.x;
  int lane = tid & 31;
  int wave = tid >> 5;                             // 0..7
  int half = lane >> 4;                            // lane group (0: lanes 0-15, 1: 16-31)
  int l16  = lane & 15;

  if (tid < MT) {
    int idx = base + tid;
    int t = (idx < count) ? tokl[e * N_TOK + idx] : -1;
    tok_s[tid] = t;
    prb_s[tid] = (idx < count) ? prbl[e * N_TOK + idx] : 0.f;
  }
  __syncthreads();
  for (int i = tid; i < MT * C_DIM; i += 256) {    // gather x rows -> LDS
    int r = i >> 10, c = i & (C_DIM - 1);
    int t = tok_s[r];
    Xs[r][c] = (t >= 0) ? xb[(size_t)t * C_DIM + c] : (unsigned short)0;
  }
  __syncthreads();

  // GEMM1 tile assignment: wave -> (m1 = wave/4 in {0,1}, n1 = wave%4 in {0..3})
  int m1 = wave >> 2, n1 = wave & 3;
  // GEMM2 assignment: wave -> (m2 = wave&1, column group ng = wave>>1, 16 n-tiles each)
  int m2 = wave & 1, ng = wave >> 1;

  v8f yacc[16];
  #pragma unroll
  for (int i = 0; i < 16; ++i) yacc[i] = (v8f){0.f,0.f,0.f,0.f,0.f,0.f,0.f,0.f};

  const unsigned short* xrowA = &Xs[m1 * 16 + l16][16 * half];
  const unsigned short* w2base =
      w2t + ((size_t)e * C_DIM + ng * 256 + l16) * H_DIM + 16 * half;
  const float* b1e = b1 + (size_t)e * H_DIM;

  for (int hc = 0; hc < H_DIM; hc += HC) {
    // ---- GEMM1: h(m1,n1) [16x16] = X[m1] @ w1[:, hc+n1*16 ..]  (K = C = 1024) ----
    v8f hacc = (v8f){0.f,0.f,0.f,0.f,0.f,0.f,0.f,0.f};
    const unsigned short* w1row =
        w1t + ((size_t)e * H_DIM + hc + n1 * 16 + l16) * C_DIM + 16 * half;
    for (int k0 = 0; k0 < C_DIM; k0 += 32) {
      v16bf av = ld_b_contig(xrowA + k0);
      v16bf bv = ld_b_contig(w1row + k0);
      hacc = __builtin_amdgcn_wmma_f32_16x16x32_bf16(
          false, av, false, bv, (short)0, hacc, false, false);
    }
    __syncthreads();                               // prev chunk's Hs fully consumed
    #pragma unroll
    for (int r = 0; r < 8; ++r) {                  // bias + relu -> bf16 -> LDS
      int row = r + 8 * half;
      float v = hacc[r] + b1e[hc + n1 * 16 + l16];
      v = v > 0.f ? v : 0.f;
      Hs[m1 * 16 + row][n1 * 16 + l16] = bf16_rne(v);
    }
    __syncthreads();
    // ---- GEMM2 partial: y[m2, ng] += Hs[m2] @ w2[hc..hc+64, ng*256..] ----
    #pragma unroll
    for (int ks = 0; ks < 2; ++ks) {
      int k0 = ks * 32;
      v16bf av = ld_b_contig(&Hs[m2 * 16 + l16][k0 + 16 * half]);
      #pragma unroll
      for (int nt = 0; nt < 16; ++nt) {
        v16bf bv = ld_b_contig(w2base + (size_t)nt * 16 * H_DIM + hc + k0);
        yacc[nt] = __builtin_amdgcn_wmma_f32_16x16x32_bf16(
            false, av, false, bv, (short)0, yacc[nt], false, false);
      }
    }
  }

  // ---- epilogue: out[token, col] += prob * (y + b2) ----
  const float* b2e = b2 + (size_t)e * C_DIM;
  #pragma unroll
  for (int nt = 0; nt < 16; ++nt) {
    int col = ng * 256 + nt * 16 + l16;
    float bb = b2e[col];
    #pragma unroll
    for (int r = 0; r < 8; ++r) {
      int row = m2 * 16 + r + 8 * half;
      int t = tok_s[row];
      if (t >= 0) {
        float v = prb_s[row] * (yacc[nt][r] + bb);
        unsafeAtomicAdd(&out[(size_t)t * C_DIM + col], v);
      }
    }
  }
}

// ---------------- host launcher ----------------
extern "C" void kernel_launch(void* const* d_in, const int* in_sizes, int n_in,
                              void* d_out, int out_size, void* d_ws, size_t ws_size,
                              hipStream_t stream) {
  (void)in_sizes; (void)n_in; (void)out_size;
  if (ws_size < WS_NEED) return;  // needs ~161 MB scratch

  const float* x   = (const float*)d_in[0];
  const float* eps = (const float*)d_in[1];
  const float* rw  = (const float*)d_in[2];
  const float* rb  = (const float*)d_in[3];
  const float* nw  = (const float*)d_in[4];
  const float* nb  = (const float*)d_in[5];
  const float* w1  = (const float*)d_in[6];
  const float* b1  = (const float*)d_in[7];
  const float* w2  = (const float*)d_in[8];
  const float* b2  = (const float*)d_in[9];
  float* out = (float*)d_out;

  char* ws = (char*)d_ws;
  int*            cnt  = (int*)(ws + OFF_CNT);
  int*            tokl = (int*)(ws + OFF_TOK);
  float*          prbl = (float*)(ws + OFF_PRB);
  unsigned short* xb   = (unsigned short*)(ws + OFF_XB);
  unsigned short* w1t  = (unsigned short*)(ws + OFF_W1T);
  unsigned short* w2t  = (unsigned short*)(ws + OFF_W2T);

  const size_t nOut = (size_t)N_TOK * C_DIM;

  moe_zero_kernel<<<4096, 256, 0, stream>>>(out, cnt, nOut);
  moe_cvt_kernel<<<4096, 256, 0, stream>>>(x, xb, nOut);
  // w1 [E][C][H] -> w1t [E][H][C]
  moe_transpose_cvt_kernel<<<dim3(H_DIM / 32, C_DIM / 32, E_NUM), 256, 0, stream>>>(
      w1, w1t, C_DIM, H_DIM);
  // w2 [E][H][C] -> w2t [E][C][H]
  moe_transpose_cvt_kernel<<<dim3(C_DIM / 32, H_DIM / 32, E_NUM), 256, 0, stream>>>(
      w2, w2t, H_DIM, C_DIM);
  moe_route_kernel<<<N_TOK / 8, 256, 0, stream>>>(x, eps, rw, rb, nw, nb,
                                                  cnt, tokl, prbl);
  moe_ffn_kernel<<<dim3(N_TOK / MT, E_NUM), 256, 0, stream>>>(
      b1, b2, xb, w1t, w2t, cnt, tokl, prbl, out);
}
